// LightGCN_Metadata_55542517071980
// MI455X (gfx1250) — matrix operand order repaired and checked by reading, not compiled, not run on hardware
//
#include <hip/hip_runtime.h>
#include <hip/hip_bf16.h>

// ---------------------------------------------------------------------------
// Problem constants (from the reference)
// ---------------------------------------------------------------------------
#define NUM_USERS 50000
#define NUM_ITEMS 20000
#define NUM_NODES 70000
#define FEAT_DIM 768
#define HID 128
#define N_EDGES 2000000
#define ALPHA 0.25f    // 1/(NUM_LAYERS+1)
#define LN_EPS 1e-5f

// ---------------------------------------------------------------------------
// WMMA types (CDNA5 / gfx1250, wave32)
// ---------------------------------------------------------------------------
typedef __attribute__((ext_vector_type(16))) __bf16 v16bf;
typedef __attribute__((ext_vector_type(8)))  float  v8f;

union FragAB { v16bf v; uint4 q[2]; };

__device__ __forceinline__ __bf16 f2bf(float f) {
  union { float f; unsigned u; } in;
  in.f = f;
  unsigned u = in.u;
  u = (u + 0x7FFFu + ((u >> 16) & 1u)) >> 16;   // round-to-nearest-even
  union { unsigned short s; __bf16 h; } out;
  out.s = (unsigned short)u;
  return out.h;
}

// Async global->LDS 16B copy (CDNA5 GLOBAL_LOAD_ASYNC_TO_LDS_B128, ASYNCcnt).
// ldsoff: per-lane LDS byte offset (low 32 bits of a generic shared pointer),
// goff:   per-lane byte offset from sgpr base (GVS mode), base: SGPR pair.
__device__ __forceinline__ void async_copy_b128(unsigned ldsoff, unsigned goff,
                                                const void* base) {
  asm volatile("global_load_async_to_lds_b128 %0, %1, %2"
               :: "v"(ldsoff), "v"(goff), "s"(base) : "memory");
}
#define WAIT_ASYNCCNT(n) asm volatile("s_wait_asynccnt %0" :: "n"(n) : "memory")

// ---------------------------------------------------------------------------
// Tiled GEMM:  C[M,N] = A[M,K] (bf16) x B[K,N] (bf16) + bias[N]  (f32 accum)
// Block = 256 threads (8 waves); block tile 128x128, K-step 32.
// Waves arranged 4(M) x 2(N); each wave owns 32x64 = 2x4 WMMA tiles.
// Double-buffered LDS filled with async global->LDS b128 copies.
// Requires: N % 128 == 0, K % 32 == 0.
// ---------------------------------------------------------------------------
#define BM 128
#define BN 128
#define KB 32
#define A_ST 40     // padded LDS row stride (bf16 elems), 16B-aligned rows
#define B_ST 136

__global__ __launch_bounds__(256)
void gemm_bias_bf16wmma(const __bf16* __restrict__ A, const __bf16* __restrict__ B,
                        const float* __restrict__ bias, float* __restrict__ C,
                        int M, int N, int K) {
  __shared__ __bf16 As[2][BM * A_ST];
  __shared__ __bf16 Bs[2][KB * B_ST];

  const int tid   = threadIdx.x;
  const int lane  = tid & 31;
  const int wid   = tid >> 5;
  const int waveM = wid & 3;    // 0..3 : which 32-row slab
  const int waveN = wid >> 2;   // 0..1 : which 64-col slab
  const int m0 = blockIdx.y * BM;
  const int n0 = blockIdx.x * BN;

  const v8f vz = {0.f,0.f,0.f,0.f,0.f,0.f,0.f,0.f};
  v8f acc[2][4];
#pragma unroll
  for (int i = 0; i < 2; i++)
#pragma unroll
    for (int j = 0; j < 4; j++) acc[i][j] = vz;

  // Async staging maps. Each wave issues 2 A + 2 B b128 instructions per tile.
  // A tile (128x32 bf16): 8 rows x 4 16B-chunks per instruction.
  const int a_r = lane >> 2;            // 0..7
  const int a_c = (lane & 3) << 3;      // elem 0,8,16,24
  // B tile (32x128 bf16): 512 16B-chunks; chunk -> (row, 8-elem col group).
  const int nk = K / KB;

  // prologue: stage tile 0 into buffer 0
  {
    const int k0 = 0;
#pragma unroll
    for (int h = 0; h < 2; h++) {
      int row  = wid * 16 + h * 8 + a_r;
      int grow = m0 + row; if (grow >= M) grow = M - 1;   // clamp, rows discarded later
      unsigned goff = (unsigned)(((size_t)grow * K + k0 + a_c) * 2);
      unsigned lo   = (unsigned)(uintptr_t)&As[0][row * A_ST + a_c];
      async_copy_b128(lo, goff, A);
    }
#pragma unroll
    for (int h = 0; h < 2; h++) {
      int chunk = wid * 64 + h * 32 + lane;
      int brow  = chunk >> 4;
      int bcol  = (chunk & 15) << 3;
      unsigned goff = (unsigned)(((size_t)(k0 + brow) * N + n0 + bcol) * 2);
      unsigned lo   = (unsigned)(uintptr_t)&Bs[0][brow * B_ST + bcol];
      async_copy_b128(lo, goff, B);
    }
  }

  const int arow = lane & 15;
  const int akb  = (lane >> 4) << 3;   // 0 or 8

  for (int kt = 0; kt < nk; kt++) {
    const int cur = kt & 1;
    if (kt + 1 < nk) {
      // stage tile kt+1 into the other buffer while we compute on `cur`
      const int k0 = (kt + 1) * KB;
      const int nxt = cur ^ 1;
#pragma unroll
      for (int h = 0; h < 2; h++) {
        int row  = wid * 16 + h * 8 + a_r;
        int grow = m0 + row; if (grow >= M) grow = M - 1;
        unsigned goff = (unsigned)(((size_t)grow * K + k0 + a_c) * 2);
        unsigned lo   = (unsigned)(uintptr_t)&As[nxt][row * A_ST + a_c];
        async_copy_b128(lo, goff, A);
      }
#pragma unroll
      for (int h = 0; h < 2; h++) {
        int chunk = wid * 64 + h * 32 + lane;
        int brow  = chunk >> 4;
        int bcol  = (chunk & 15) << 3;
        unsigned goff = (unsigned)(((size_t)(k0 + brow) * N + n0 + bcol) * 2);
        unsigned lo   = (unsigned)(uintptr_t)&Bs[nxt][brow * B_ST + bcol];
        async_copy_b128(lo, goff, B);
      }
      WAIT_ASYNCCNT(4);   // retire tile kt's 4 copies (in-order completion)
    } else {
      WAIT_ASYNCCNT(0);
    }
    __syncthreads();      // tile kt visible from every wave

    // ---- A fragments: 16x32 bf16, ISA layout:
    //   lanes 0-15 : M=lane,    K = {0..7, 16..23}
    //   lanes 16-31: M=lane-16, K = {8..15, 24..31}
    FragAB fa[2];
#pragma unroll
    for (int mi = 0; mi < 2; mi++) {
      const __bf16* ap = &As[cur][(waveM * 32 + mi * 16 + arow) * A_ST + akb];
      fa[mi].q[0] = *(const uint4*)(ap);
      fa[mi].q[1] = *(const uint4*)(ap + 16);
    }
    // ---- B fragments: 32x16 bf16, N = lane%16, K = {akb..akb+7, 16+akb..23+akb} ----
#pragma unroll
    for (int ni = 0; ni < 4; ni++) {
      FragAB fb;
      const __bf16* bp = &Bs[cur][akb * B_ST + waveN * 64 + ni * 16 + arow];
#pragma unroll
      for (int s = 0; s < 8; s++) {
        fb.v[s]     = bp[s * B_ST];
        fb.v[8 + s] = bp[(16 + s) * B_ST];
      }
#pragma unroll
      for (int mi = 0; mi < 2; mi++) {
        acc[mi][ni] = __builtin_amdgcn_wmma_f32_16x16x32_bf16(
            false, fa[mi].v, false, fb.v, (short)0, acc[mi][ni], false, false);
      }
    }
    __syncthreads();      // everyone done reading buffer `cur` before overwrite
  }

  // ---- epilogue: C/D layout — VGPR r: M = r + 8*(lane/16), N = lane%16 ----
  const int crow = 8 * (lane >> 4);
  const int ccol = lane & 15;
#pragma unroll
  for (int mi = 0; mi < 2; mi++) {
#pragma unroll
    for (int ni = 0; ni < 4; ni++) {
      const int gcol = n0 + waveN * 64 + ni * 16 + ccol;
      const float bv = bias[gcol];
#pragma unroll
      for (int r = 0; r < 8; r++) {
        const int grow = m0 + waveM * 32 + mi * 16 + crow + r;
        if (grow < M) C[(size_t)grow * N + gcol] = acc[mi][ni][r] + bv;
      }
    }
  }
}

// ---------------------------------------------------------------------------
// f32 -> bf16 conversion prepass (4 elems / thread)
// ---------------------------------------------------------------------------
__global__ void cvt_bf16_kernel(const float* __restrict__ src,
                                __bf16* __restrict__ dst, int n) {
  int base = (blockIdx.x * blockDim.x + threadIdx.x) * 4;
  if (base + 3 < n) {
    float4 v = *(const float4*)(src + base);
    union { __bf16 b[4]; uint2 q; } pk;
    pk.b[0] = f2bf(v.x); pk.b[1] = f2bf(v.y); pk.b[2] = f2bf(v.z); pk.b[3] = f2bf(v.w);
    *(uint2*)(dst + base) = pk.q;
  } else {
    for (int j = base; j < n; j++) dst[j] = f2bf(src[j]);
  }
}

// ---------------------------------------------------------------------------
// LayerNorm + ReLU, output as bf16. One wave per row (wave32 shuffle reduce).
// ---------------------------------------------------------------------------
__global__ __launch_bounds__(256)
void ln_relu_bf16_kernel(const float* __restrict__ x, const float* __restrict__ g,
                         const float* __restrict__ b, __bf16* __restrict__ y,
                         int rows, int cols) {
  const int lane = threadIdx.x & 31;
  const int wid  = threadIdx.x >> 5;
  const int row  = blockIdx.x * 8 + wid;
  if (row >= rows) return;
  const float* p = x + (size_t)row * cols;
  __bf16* q = y + (size_t)row * cols;
  float s = 0.f, ss = 0.f;
  for (int c = lane; c < cols; c += 32) { float v = p[c]; s += v; ss += v * v; }
#pragma unroll
  for (int o = 16; o > 0; o >>= 1) {
    s  += __shfl_xor(s,  o, 32);
    ss += __shfl_xor(ss, o, 32);
  }
  const float inv_n = 1.0f / (float)cols;
  const float mu  = s * inv_n;
  const float var = ss * inv_n - mu * mu;
  const float inv = rsqrtf(var + LN_EPS);
  for (int c = lane; c < cols; c += 32) {
    float v = (p[c] - mu) * inv * g[c] + b[c];
    q[c] = f2bf(v > 0.f ? v : 0.f);
  }
}

// ---------------------------------------------------------------------------
// Fuse: L2-normalize item metadata, build e0, seed x0 = e0 and out = alpha*e0
// ---------------------------------------------------------------------------
__global__ __launch_bounds__(256)
void fuse_kernel(const float* __restrict__ emb, const float* __restrict__ h3,
                 const float* __restrict__ mw, float* __restrict__ x0,
                 float* __restrict__ out) {
  const int lane = threadIdx.x & 31;
  const int wid  = threadIdx.x >> 5;
  const int row  = blockIdx.x * 8 + wid;
  if (row >= NUM_NODES) return;
  float4 e = ((const float4*)emb)[(size_t)row * 32 + lane];
  if (row >= NUM_USERS) {
    const int i = row - NUM_USERS;
    float4 h = ((const float4*)h3)[(size_t)i * 32 + lane];
    float ss = h.x*h.x + h.y*h.y + h.z*h.z + h.w*h.w;
#pragma unroll
    for (int o = 16; o > 0; o >>= 1) ss += __shfl_xor(ss, o, 32);
    const float w = mw[0] / fmaxf(sqrtf(ss), 1e-12f);
    e.x += w * h.x; e.y += w * h.y; e.z += w * h.z; e.w += w * h.w;
  }
  ((float4*)x0)[(size_t)row * 32 + lane] = e;
  float4 o4;
  o4.x = ALPHA * e.x; o4.y = ALPHA * e.y; o4.z = ALPHA * e.z; o4.w = ALPHA * e.w;
  ((float4*)out)[(size_t)row * 32 + lane] = o4;
}

// ---------------------------------------------------------------------------
// Graph propagation helpers
// ---------------------------------------------------------------------------
__global__ void fill_zero_kernel(float* __restrict__ p, int n) {
  int i = blockIdx.x * blockDim.x + threadIdx.x;
  if (i < n) p[i] = 0.f;
}

__global__ void degree_kernel(const int* __restrict__ ei, float* __restrict__ deg) {
  int e = blockIdx.x * blockDim.x + threadIdx.x;
  if (e < N_EDGES) atomicAdd(&deg[ei[N_EDGES + e]], 1.0f);
}

__global__ void dinv_kernel(const float* __restrict__ deg, float* __restrict__ dinv, int n) {
  int i = blockIdx.x * blockDim.x + threadIdx.x;
  if (i < n) {
    float d = deg[i];
    dinv[i] = d > 0.f ? rsqrtf(fmaxf(d, 1.0f)) : 0.f;
  }
}

// one wave per edge: gather x[src] (float4/lane), scatter-add into xn[dst]
__global__ __launch_bounds__(256)
void scatter_kernel(const int* __restrict__ ei, const float* __restrict__ dinv,
                    const float* __restrict__ x, float* __restrict__ xn) {
  const int lane = threadIdx.x & 31;
  const int wid  = threadIdx.x >> 5;
  const int e    = blockIdx.x * 8 + wid;
  if (e >= N_EDGES) return;
  const int s = ei[e];
  const int d = ei[N_EDGES + e];
  const float c = dinv[s] * dinv[d];
  float4 v = ((const float4*)x)[(size_t)s * 32 + lane];
  float* dp = xn + (size_t)d * HID + lane * 4;
  atomicAdd(dp + 0, c * v.x);
  atomicAdd(dp + 1, c * v.y);
  atomicAdd(dp + 2, c * v.z);
  atomicAdd(dp + 3, c * v.w);
}

__global__ void axpy_kernel(float* __restrict__ out, const float* __restrict__ xn, int n) {
  int i = blockIdx.x * blockDim.x + threadIdx.x;
  if (i < n) out[i] += ALPHA * xn[i];
}

// ---------------------------------------------------------------------------
// Orchestration
// ---------------------------------------------------------------------------
extern "C" void kernel_launch(void* const* d_in, const int* in_sizes, int n_in,
                              void* d_out, int out_size, void* d_ws, size_t ws_size,
                              hipStream_t stream) {
  const int*   ei   = (const int*)  d_in[0];
  const float* feat = (const float*)d_in[1];
  const float* emb  = (const float*)d_in[2];
  const float* W1   = (const float*)d_in[3];
  const float* b1   = (const float*)d_in[4];
  const float* g1   = (const float*)d_in[5];
  const float* be1  = (const float*)d_in[6];
  const float* W2   = (const float*)d_in[7];
  const float* b2   = (const float*)d_in[8];
  const float* g2   = (const float*)d_in[9];
  const float* be2  = (const float*)d_in[10];
  const float* W3   = (const float*)d_in[11];
  const float* b3   = (const float*)d_in[12];
  const float* mw   = (const float*)d_in[13];
  float* out = (float*)d_out;

  // Workspace layout (bytes, all offsets 256-aligned). All regions are fully
  // rewritten every call before use, so aliasing is replay-safe.
  char* ws = (char*)d_ws;
  __bf16* Abf  = (__bf16*)(ws + 0);             // 20000*768*2  = 30,720,000
  __bf16* W1bf = (__bf16*)(ws + 30720000);      // 768*512*2    =    786,432
  __bf16* H1bf = (__bf16*)(ws + 31506432);      // 20000*512*2  = 20,480,000
  float*  H1f  = (float*) (ws + 51986432);      // 20000*512*4  = 40,960,000
  float*  xA   = (float*) (ws + 51986432);      // alias of H1f (dead after ln1)
  float*  H2f  = (float*) (ws + 92946432);      // 20000*128*4  = 10,240,000
  __bf16* H2bf = (__bf16*)(ws + 103186432);     // 20000*128*2  =  5,120,000
  float*  H3   = (float*) (ws + 108306432);     // 20000*128*4  = 10,240,000
  __bf16* W2bf = (__bf16*)(ws + 118546432);     // 512*128*2    =    131,072
  __bf16* W3bf = (__bf16*)(ws + 118677504);     // 128*128*2    =     32,768
  float*  deg  = (float*) (ws + 118710272);     // 70000*4 (pad 280,064)
  float*  dnv  = (float*) (ws + 118990336);     // 70000*4 (pad 280,064)
  float*  xB   = (float*) (ws + 0);             // alias Abf/W1bf/H1bf (dead after GEMM2)
  // total ws required: 119,270,400 bytes

  // ---- bf16 conversion prepasses (bandwidth-bound, one-shot) ----
  cvt_bf16_kernel<<<(NUM_ITEMS * FEAT_DIM / 4 + 255) / 256, 256, 0, stream>>>(
      feat, Abf, NUM_ITEMS * FEAT_DIM);
  cvt_bf16_kernel<<<(FEAT_DIM * 512 / 4 + 255) / 256, 256, 0, stream>>>(
      W1, W1bf, FEAT_DIM * 512);
  cvt_bf16_kernel<<<(512 * HID / 4 + 255) / 256, 256, 0, stream>>>(
      W2, W2bf, 512 * HID);
  cvt_bf16_kernel<<<(HID * HID / 4 + 255) / 256, 256, 0, stream>>>(
      W3, W3bf, HID * HID);

  // ---- item metadata MLP (async-staged bf16 WMMA GEMMs) ----
  gemm_bias_bf16wmma<<<dim3(512 / BN, (NUM_ITEMS + BM - 1) / BM), 256, 0, stream>>>(
      Abf, W1bf, b1, H1f, NUM_ITEMS, 512, FEAT_DIM);
  ln_relu_bf16_kernel<<<(NUM_ITEMS + 7) / 8, 256, 0, stream>>>(
      H1f, g1, be1, H1bf, NUM_ITEMS, 512);

  gemm_bias_bf16wmma<<<dim3(1, (NUM_ITEMS + BM - 1) / BM), 256, 0, stream>>>(
      H1bf, W2bf, b2, H2f, NUM_ITEMS, HID, 512);
  ln_relu_bf16_kernel<<<(NUM_ITEMS + 7) / 8, 256, 0, stream>>>(
      H2f, g2, be2, H2bf, NUM_ITEMS, HID);

  gemm_bias_bf16wmma<<<dim3(1, (NUM_ITEMS + BM - 1) / BM), 256, 0, stream>>>(
      H2bf, W3bf, b3, H3, NUM_ITEMS, HID, HID);

  // ---- fuse metadata, seed x0 and out ----
  fuse_kernel<<<(NUM_NODES + 7) / 8, 256, 0, stream>>>(emb, H3, mw, xA, out);

  // ---- symmetric GCN normalization ----
  fill_zero_kernel<<<(NUM_NODES + 255) / 256, 256, 0, stream>>>(deg, NUM_NODES);
  degree_kernel<<<(N_EDGES + 255) / 256, 256, 0, stream>>>(ei, deg);
  dinv_kernel<<<(NUM_NODES + 255) / 256, 256, 0, stream>>>(deg, dnv, NUM_NODES);

  // ---- 3 propagation layers (state fits in 192MB L2 -> L2-atomic bound) ----
  const int nstate = NUM_NODES * HID;   // 8,960,000
  float* xc = xA;
  float* xn = xB;
  for (int l = 0; l < 3; l++) {
    fill_zero_kernel<<<(nstate + 255) / 256, 256, 0, stream>>>(xn, nstate);
    scatter_kernel<<<(N_EDGES + 7) / 8, 256, 0, stream>>>(ei, dnv, xc, xn);
    axpy_kernel<<<(nstate + 255) / 256, 256, 0, stream>>>(out, xn, nstate);
    float* t = xc; xc = xn; xn = t;
  }
}
